// _FusedTrellisExpert_7808250544398
// MI455X (gfx1250) — compile-verified, matrix operand order checked by reading, not run on hardware
//
#include <hip/hip_runtime.h>

// MI455X / gfx1250 fused SwiGLU MLP, bf16 WMMA pipeline:
//   K0 (x3): fp32 -> bf16 pre-conversion (memory-bound, ~13us at HBM roofline)
//   K1: gate_up GEMM + fused SwiGLU -> h (bf16) ; async global->LDS staging
//   K2: down GEMM -> out (f32)               ; async global->LDS staging
// Workspace: h 92MB + x_bf 32MB + wgu_bf 44MB + wd_bf 22MB  (~186MB)

#define HIDDEN 2048
#define INTER  5632
#define TOKENS 8192

typedef __attribute__((ext_vector_type(16))) __bf16 v16bf;
typedef __attribute__((ext_vector_type(8)))  float  v8f;

__device__ __forceinline__ unsigned short f2bf(float f) {
    union { float f; unsigned int u; } c; c.f = f;
    unsigned int u = c.u + 0x7FFFu + ((c.u >> 16) & 1u);   // round-to-nearest-even
    return (unsigned short)(u >> 16);
}

// Per-lane async copy: 16 bytes global -> LDS, tracked by ASYNCcnt.
__device__ __forceinline__ void async_copy16(unsigned int lds_off,
                                             const unsigned short* gptr) {
    asm volatile("global_load_async_to_lds_b128 %0, %1, off"
                 :: "v"(lds_off), "v"((unsigned long long)(uintptr_t)gptr)
                 : "memory");
}
__device__ __forceinline__ void wait_async0() {
    asm volatile("s_wait_asynccnt 0x0" ::: "memory");
}

// ---------------------------------------------------------------------------
// K0: elementwise fp32 -> bf16, 4 elements per thread (sizes divide exactly).
// ---------------------------------------------------------------------------
__global__ __launch_bounds__(256) void k0_convert(
    const float* __restrict__ src, unsigned short* __restrict__ dst,
    unsigned int n4)
{
    unsigned int i = blockIdx.x * 256u + threadIdx.x;
    if (i < n4) {
        float4 v = ((const float4*)src)[i];
        ushort4 o;
        o.x = f2bf(v.x); o.y = f2bf(v.y); o.z = f2bf(v.z); o.w = f2bf(v.w);
        ((ushort4*)dst)[i] = o;
    }
}

// ---------------------------------------------------------------------------
// K1: gate/up projection + SwiGLU. Block tile: 128 tokens x 64 inter-cols.
// lds_b rows 0-63 = gate weight rows [n0..), rows 64-127 = up rows [INTER+n0..)
// Waves 0-3 accumulate gate, waves 4-7 accumulate up (same 64 columns).
// ---------------------------------------------------------------------------
__global__ __launch_bounds__(256) void k1_gateup_swiglu(
    const unsigned short* __restrict__ xb,
    const unsigned short* __restrict__ wb,
    unsigned short* __restrict__ h)
{
    __shared__ __align__(32) unsigned short lds_a[128 * 32];
    __shared__ __align__(32) unsigned short lds_b[128 * 32];
    __shared__ __align__(16) float          xch[128 * 64];

    const int tid  = threadIdx.x;
    const int lane = tid & 31;
    const int wave = tid >> 5;
    const int m0   = blockIdx.y * 128;
    const int n0   = blockIdx.x * 64;

    const int  wr      = wave & 3;
    const bool is_gate = (wave < 4);
    const int  lm      = lane & 15;
    const int  lh      = lane >> 4;

    v8f acc[2][4];
    v8f vzero = {};
    #pragma unroll
    for (int mi = 0; mi < 2; ++mi)
        #pragma unroll
        for (int ni = 0; ni < 4; ++ni) acc[mi][ni] = vzero;

    // Staging: 512 chunks of 16B; thread owns chunk tid (row r0 in 0..63)
    // and chunk tid+256 (row r0+64). q = 16B quarter within the 64B row.
    const int r0 = tid >> 2;
    const int q  = tid & 3;

    const unsigned int lbA = (unsigned int)(uintptr_t)lds_a + (unsigned)(r0 * 64 + q * 16);
    const unsigned int lbB = (unsigned int)(uintptr_t)lds_b + (unsigned)(r0 * 64 + q * 16);
    const unsigned int rowStep = 64 * 64;   // 64 rows * 64B

    const unsigned short* gA0 = xb + (size_t)(m0 + r0) * HIDDEN + q * 8;
    const unsigned short* gA1 = gA0 + (size_t)64 * HIDDEN;
    const unsigned short* gB0 = wb + (size_t)(n0 + r0) * HIDDEN + q * 8;           // gate rows
    const unsigned short* gB1 = wb + (size_t)(INTER + n0 + r0) * HIDDEN + q * 8;   // up rows

    for (int k0 = 0; k0 < HIDDEN; k0 += 32) {
        async_copy16(lbA,           gA0 + k0);
        async_copy16(lbA + rowStep, gA1 + k0);
        async_copy16(lbB,           gB0 + k0);
        async_copy16(lbB + rowStep, gB1 + k0);
        wait_async0();
        __syncthreads();

        v16bf afrag[2];
        #pragma unroll
        for (int mi = 0; mi < 2; ++mi)
            afrag[mi] = *(const v16bf*)&lds_a[(wr * 32 + mi * 16 + lm) * 32 + lh * 16];

        const int bbase = is_gate ? 0 : 64;
        v16bf bfrag[4];
        #pragma unroll
        for (int ni = 0; ni < 4; ++ni)
            bfrag[ni] = *(const v16bf*)&lds_b[(bbase + ni * 16 + lm) * 32 + lh * 16];

        #pragma unroll
        for (int mi = 0; mi < 2; ++mi)
            #pragma unroll
            for (int ni = 0; ni < 4; ++ni)
                acc[mi][ni] = __builtin_amdgcn_wmma_f32_16x16x32_bf16(
                    false, afrag[mi], false, bfrag[ni],
                    (short)0, acc[mi][ni], false, false);

        __syncthreads();
    }

    // SwiGLU: up waves publish accumulators via LDS, gate waves combine+store.
    if (!is_gate) {
        #pragma unroll
        for (int mi = 0; mi < 2; ++mi)
            #pragma unroll
            for (int ni = 0; ni < 4; ++ni)
                #pragma unroll
                for (int r = 0; r < 8; ++r)
                    xch[(wr * 32 + mi * 16 + r + 8 * lh) * 64 + ni * 16 + lm] = acc[mi][ni][r];
    }
    __syncthreads();
    if (is_gate) {
        #pragma unroll
        for (int mi = 0; mi < 2; ++mi)
            #pragma unroll
            for (int ni = 0; ni < 4; ++ni)
                #pragma unroll
                for (int r = 0; r < 8; ++r) {
                    int rloc = wr * 32 + mi * 16 + r + 8 * lh;
                    float g = acc[mi][ni][r];
                    float u = xch[rloc * 64 + ni * 16 + lm];
                    float s = g / (1.0f + __expf(-g));   // silu
                    h[(size_t)(m0 + rloc) * INTER + (n0 + ni * 16 + lm)] = f2bf(s * u);
                }
    }
}

// ---------------------------------------------------------------------------
// K2: down projection. Block tile 128 tokens x 128 out-cols.
// ---------------------------------------------------------------------------
__global__ __launch_bounds__(256) void k2_down(
    const unsigned short* __restrict__ h,
    const unsigned short* __restrict__ wdb,
    float* __restrict__ out)
{
    __shared__ __align__(32) unsigned short lds_a[128 * 32];
    __shared__ __align__(32) unsigned short lds_b[128 * 32];

    const int tid  = threadIdx.x;
    const int lane = tid & 31;
    const int wave = tid >> 5;
    const int m0   = blockIdx.y * 128;
    const int n0   = blockIdx.x * 128;

    const int wr = wave & 3;
    const int wc = wave >> 2;
    const int lm = lane & 15;
    const int lh = lane >> 4;

    v8f acc[2][4];
    v8f vzero = {};
    #pragma unroll
    for (int mi = 0; mi < 2; ++mi)
        #pragma unroll
        for (int ni = 0; ni < 4; ++ni) acc[mi][ni] = vzero;

    const int r0 = tid >> 2;
    const int q  = tid & 3;

    const unsigned int lbA = (unsigned int)(uintptr_t)lds_a + (unsigned)(r0 * 64 + q * 16);
    const unsigned int lbB = (unsigned int)(uintptr_t)lds_b + (unsigned)(r0 * 64 + q * 16);
    const unsigned int rowStep = 64 * 64;

    const unsigned short* gA0 = h + (size_t)(m0 + r0) * INTER + q * 8;
    const unsigned short* gA1 = gA0 + (size_t)64 * INTER;
    const unsigned short* gB0 = wdb + (size_t)(n0 + r0) * INTER + q * 8;
    const unsigned short* gB1 = gB0 + (size_t)64 * INTER;

    for (int k0 = 0; k0 < INTER; k0 += 32) {
        async_copy16(lbA,           gA0 + k0);
        async_copy16(lbA + rowStep, gA1 + k0);
        async_copy16(lbB,           gB0 + k0);
        async_copy16(lbB + rowStep, gB1 + k0);
        wait_async0();
        __syncthreads();

        v16bf afrag[2];
        #pragma unroll
        for (int mi = 0; mi < 2; ++mi)
            afrag[mi] = *(const v16bf*)&lds_a[(wr * 32 + mi * 16 + lm) * 32 + lh * 16];

        v16bf bfrag[4];
        #pragma unroll
        for (int ni = 0; ni < 4; ++ni)
            bfrag[ni] = *(const v16bf*)&lds_b[(wc * 64 + ni * 16 + lm) * 32 + lh * 16];

        #pragma unroll
        for (int mi = 0; mi < 2; ++mi)
            #pragma unroll
            for (int ni = 0; ni < 4; ++ni)
                acc[mi][ni] = __builtin_amdgcn_wmma_f32_16x16x32_bf16(
                    false, afrag[mi], false, bfrag[ni],
                    (short)0, acc[mi][ni], false, false);

        __syncthreads();
    }

    #pragma unroll
    for (int mi = 0; mi < 2; ++mi)
        #pragma unroll
        for (int ni = 0; ni < 4; ++ni)
            #pragma unroll
            for (int r = 0; r < 8; ++r) {
                int row = m0 + wr * 32 + mi * 16 + r + 8 * lh;
                int col = n0 + wc * 64 + ni * 16 + lm;
                out[(size_t)row * HIDDEN + col] = acc[mi][ni][r];
            }
}

extern "C" void kernel_launch(void* const* d_in, const int* in_sizes, int n_in,
                              void* d_out, int out_size, void* d_ws, size_t ws_size,
                              hipStream_t stream) {
    const float* x   = (const float*)d_in[0];   // [TOKENS, HIDDEN]
    const float* wgu = (const float*)d_in[1];   // [2*INTER, HIDDEN]
    const float* wd  = (const float*)d_in[2];   // [HIDDEN, INTER]
    float* out = (float*)d_out;                 // [TOKENS, HIDDEN]

    // Workspace carve-up (all sizes multiples of 4K elements).
    unsigned short* hbuf = (unsigned short*)d_ws;                  // TOKENS*INTER
    unsigned short* xb   = hbuf + (size_t)TOKENS * INTER;          // TOKENS*HIDDEN
    unsigned short* wgub = xb   + (size_t)TOKENS * HIDDEN;         // 2*INTER*HIDDEN
    unsigned short* wdb  = wgub + (size_t)2 * INTER * HIDDEN;      // HIDDEN*INTER

    const unsigned int nx  = (unsigned int)((size_t)TOKENS * HIDDEN / 4);
    const unsigned int ngu = (unsigned int)((size_t)2 * INTER * HIDDEN / 4);
    const unsigned int nwd = (unsigned int)((size_t)HIDDEN * INTER / 4);

    k0_convert<<<dim3(nx  / 256), dim3(256), 0, stream>>>(x,   xb,   nx);
    k0_convert<<<dim3(ngu / 256), dim3(256), 0, stream>>>(wgu, wgub, ngu);
    k0_convert<<<dim3(nwd / 256), dim3(256), 0, stream>>>(wd,  wdb,  nwd);

    dim3 g1(INTER / 64, TOKENS / 128);   // 88 x 64
    k1_gateup_swiglu<<<g1, dim3(256), 0, stream>>>(xb, wgub, hbuf);

    dim3 g2(HIDDEN / 128, TOKENS / 128); // 16 x 64
    k2_down<<<g2, dim3(256), 0, stream>>>(hbuf, wdb, out);
}
